// GIN_3633542332749
// MI455X (gfx1250) — matrix operand order repaired and checked by reading, not compiled, not run on hardware
//
#include <hip/hip_runtime.h>
#include <hip/hip_bf16.h>

#define NN 50000
#define EE 600000
#define DD 128
#define LL 4
#define CC 10
#define GG 512

typedef __attribute__((ext_vector_type(16))) _Float16 v16h;
typedef __attribute__((ext_vector_type(8)))  float    v8f;

__device__ __forceinline__ float4 f4add(float4 a, float4 b) {
    return make_float4(a.x + b.x, a.y + b.y, a.z + b.z, a.w + b.w);
}
__device__ __forceinline__ float4 f4bnrelu(float4 v, float4 sc, float4 sh) {
    float4 r;
    r.x = fmaxf(v.x * sc.x + sh.x, 0.0f);
    r.y = fmaxf(v.y * sc.y + sh.y, 0.0f);
    r.z = fmaxf(v.z * sc.z + sh.z, 0.0f);
    r.w = fmaxf(v.w * sc.w + sh.w, 0.0f);
    return r;
}
__device__ __forceinline__ void pack8(v16h& a, int base, float4 u0, float4 u1) {
    a[base + 0] = (_Float16)u0.x; a[base + 1] = (_Float16)u0.y;
    a[base + 2] = (_Float16)u0.z; a[base + 3] = (_Float16)u0.w;
    a[base + 4] = (_Float16)u1.x; a[base + 5] = (_Float16)u1.y;
    a[base + 6] = (_Float16)u1.z; a[base + 7] = (_Float16)u1.w;
}

// ---------------------------------------------------------------------------
// Convert + transpose conv weights to f16:  Wt[i][n][k] = (half)W[i][k][n]
// ---------------------------------------------------------------------------
__global__ void wconv_kernel(const float* __restrict__ Wc1,
                             const float* __restrict__ Wc2,
                             _Float16* __restrict__ W1t,
                             _Float16* __restrict__ W2t) {
    int t = blockIdx.x * blockDim.x + threadIdx.x;       // L*D*D = 65536
    if (t >= LL * DD * DD) return;
    int i   = t >> 14;
    int rem = t & 16383;
    int k   = rem >> 7;
    int n   = rem & 127;
    int dstIdx = (i << 14) + n * DD + k;
    W1t[dstIdx] = (_Float16)Wc1[t];
    W2t[dstIdx] = (_Float16)Wc2[t];
}

// ---------------------------------------------------------------------------
// Edge scatter-add: agg[dst] += h[src].  32 lanes per edge, float4 per lane.
// ---------------------------------------------------------------------------
__global__ void agg_kernel(const float* __restrict__ h,
                           const long long* __restrict__ src,
                           const long long* __restrict__ dst,
                           float* __restrict__ agg) {
    int t    = blockIdx.x * blockDim.x + threadIdx.x;
    int e    = t >> 5;
    int lane = t & 31;
    if (e >= EE) return;
    int s = (int)src[e];
    int d = (int)dst[e];
    const float4 v = *(const float4*)(h + (size_t)s * DD + lane * 4);
    float* out = agg + (size_t)d * DD + lane * 4;
    atomicAdd(out + 0, v.x);
    atomicAdd(out + 1, v.y);
    atomicAdd(out + 2, v.z);
    atomicAdd(out + 3, v.w);
}

// ---------------------------------------------------------------------------
// WMMA row-panel GEMM core.  One wave computes rows [tm, tm+16) x all 128
// columns: 8 accumulator tiles, A fragment built once per k-block and reused
// for 8 WMMAs.  Bt is f16 pre-transposed weight: Bt[n][k].
//
// A fragment layout (16-bit A 16x32): lanes 0-15 row M=lane, halves [0..15] =
// K=k0..k0+15 with k0 = kb; lanes 16-31 use k0 = kb+16.
// B fragment (32x16): lanes 0-15 col N=lane hold K=kb..kb+15 consecutively,
// lanes 16-31 hold K=kb+16..kb+31.
// C/D f32: VGPR r -> row r + 8*(lane>=16), col lane&15.
// ---------------------------------------------------------------------------
template <int MODE>
__device__ __forceinline__ void gemm_panel(const float* __restrict__ A0,
                                           const float* __restrict__ A1,
                                           const float* __restrict__ scale,
                                           const float* __restrict__ shift,
                                           const _Float16* __restrict__ Bt,
                                           const float* __restrict__ bias,
                                           float* __restrict__ Cout) {
    int wave    = threadIdx.x >> 5;
    int rowTile = blockIdx.x * 4 + wave;                 // NN/16 = 3125 tiles
    if (rowTile >= NN / 16) return;
    int lane     = threadIdx.x & 31;
    int half_sel = lane >> 4;
    int tm       = rowTile * 16;
    int mrow     = tm + (lane & 15);
    int ncol_lo  = lane & 15;

    v8f acc[8];
    #pragma unroll
    for (int tn = 0; tn < 8; ++tn) acc[tn] = (v8f){};

    const float4* arow4 = (const float4*)(A0 + (size_t)mrow * DD);

    #pragma unroll
    for (int kb = 0; kb < 4; ++kb) {
        int k0 = kb * 32 + half_sel * 16;                // float index
        int f  = k0 >> 2;                                // float4 index
        v16h a;
        if (MODE == 0) {
            const float4* brow4 = (const float4*)(A1 + (size_t)mrow * DD);
            float4 x0 = f4add(arow4[f],     brow4[f]);
            float4 x1 = f4add(arow4[f + 1], brow4[f + 1]);
            float4 x2 = f4add(arow4[f + 2], brow4[f + 2]);   // K + 8
            float4 x3 = f4add(arow4[f + 3], brow4[f + 3]);
            pack8(a, 0, x0, x1);
            pack8(a, 8, x2, x3);
        } else {
            const float4* sc4 = (const float4*)scale;
            const float4* sh4 = (const float4*)shift;
            float4 x0 = f4bnrelu(arow4[f],     sc4[f],     sh4[f]);
            float4 x1 = f4bnrelu(arow4[f + 1], sc4[f + 1], sh4[f + 1]);
            float4 x2 = f4bnrelu(arow4[f + 2], sc4[f + 2], sh4[f + 2]);
            float4 x3 = f4bnrelu(arow4[f + 3], sc4[f + 3], sh4[f + 3]);
            pack8(a, 0, x0, x1);
            pack8(a, 8, x2, x3);
        }
        int kbb = kb * 32 + half_sel * 16;
        #pragma unroll
        for (int tn = 0; tn < 8; ++tn) {
            v16h b = *(const v16h*)(Bt + (size_t)(tn * 16 + ncol_lo) * DD + kbb);
            acc[tn] = __builtin_amdgcn_wmma_f32_16x16x32_f16(
                false, a, false, b, (short)0, acc[tn], false, false);
        }
    }

    #pragma unroll
    for (int tn = 0; tn < 8; ++tn) {
        int ncol = tn * 16 + ncol_lo;
        float bv = bias[ncol];
        #pragma unroll
        for (int r = 0; r < 8; ++r) {
            int row = tm + r + half_sel * 8;
            Cout[(size_t)row * DD + ncol] = acc[tn][r] + bv;
        }
    }
}

// mode 0: A = A0 + A1 (h + agg)
__global__ void gemm_add_kernel(const float* __restrict__ A0,
                                const float* __restrict__ A1,
                                const _Float16* __restrict__ Bt,
                                const float* __restrict__ bias,
                                float* __restrict__ Cout) {
    gemm_panel<0>(A0, A1, nullptr, nullptr, Bt, bias, Cout);
}

// mode 1: A = relu(A0*scale + shift)  (BN+ReLU fused into A fragment)
__global__ void gemm_bn_kernel(const float* __restrict__ A0,
                               const float* __restrict__ scale,
                               const float* __restrict__ shift,
                               const _Float16* __restrict__ Bt,
                               const float* __restrict__ bias,
                               float* __restrict__ Cout) {
    gemm_panel<1>(A0, nullptr, scale, shift, Bt, bias, Cout);
}

// ---------------------------------------------------------------------------
// BN statistics: per-column sum / sumsq over a 128-row panel, LDS fold,
// one atomicAdd pair per column per block.  sums[0..127]=sum, [128..255]=sumsq
// ---------------------------------------------------------------------------
__global__ void stats_kernel(const float* __restrict__ t,
                             float* __restrict__ sums) {
    __shared__ float ls[256];
    __shared__ float lss[256];
    int col  = threadIdx.x & 127;
    int rofs = threadIdx.x >> 7;
    int base = blockIdx.x * 128;
    float s = 0.0f, ss = 0.0f;
    for (int r = base + rofs; r < base + 128 && r < NN; r += 2) {
        float v = t[(size_t)r * DD + col];
        s  += v;
        ss += v * v;
    }
    ls[threadIdx.x]  = s;
    lss[threadIdx.x] = ss;
    __syncthreads();
    if (threadIdx.x < 128) {
        atomicAdd(&sums[col],       ls[threadIdx.x] + ls[threadIdx.x + 128]);
        atomicAdd(&sums[128 + col], lss[threadIdx.x] + lss[threadIdx.x + 128]);
    }
}

// ---------------------------------------------------------------------------
// Finalize BN: scale = gamma*rsqrt(var+eps), shift = beta - mu*scale
// ---------------------------------------------------------------------------
__global__ void bnfin_kernel(const float* __restrict__ sums,
                             const float* __restrict__ gamma,
                             const float* __restrict__ beta,
                             float* __restrict__ scale,
                             float* __restrict__ shift) {
    int d = threadIdx.x;                                 // 128 threads
    const float invN = 1.0f / (float)NN;
    float mu  = sums[d] * invN;
    float var = sums[128 + d] * invN - mu * mu;
    float sc  = gamma[d] * rsqrtf(var + 1e-5f);
    scale[d] = sc;
    shift[d] = beta[d] - mu * sc;
}

// ---------------------------------------------------------------------------
// Elementwise: h_out = relu(u*scale[d] + shift[d])
// ---------------------------------------------------------------------------
__global__ void apply_kernel(const float* __restrict__ u,
                             const float* __restrict__ scale,
                             const float* __restrict__ shift,
                             float* __restrict__ hout) {
    int t = blockIdx.x * blockDim.x + threadIdx.x;
    if (t >= NN * DD) return;
    int d = t & 127;
    float v = u[t] * scale[d] + shift[d];
    hout[t] = v > 0.0f ? v : 0.0f;
}

// ---------------------------------------------------------------------------
// Graph segment ranges (batch is sorted): start/end per graph via atomics
// ---------------------------------------------------------------------------
__global__ void range_init_kernel(int* __restrict__ start, int* __restrict__ end) {
    int g = blockIdx.x * blockDim.x + threadIdx.x;
    if (g < GG) { start[g] = NN; end[g] = 0; }
}

__global__ void range_kernel(const long long* __restrict__ batch,
                             int* __restrict__ start, int* __restrict__ end) {
    int i = blockIdx.x * blockDim.x + threadIdx.x;
    if (i >= NN) return;
    int b = (int)batch[i];
    atomicMin(&start[b], i);
    atomicMax(&end[b], i + 1);
}

// ---------------------------------------------------------------------------
// Per-graph pooling: pooled[g][d] = sum over contiguous node range
// ---------------------------------------------------------------------------
__global__ void pool_kernel(const float* __restrict__ h,
                            const int* __restrict__ start,
                            const int* __restrict__ end,
                            float* __restrict__ pooled) {
    int g = blockIdx.x;
    int d = threadIdx.x;                                 // 128 threads
    float s = 0.0f;
    int e = end[g];
    for (int r = start[g]; r < e; ++r) s += h[(size_t)r * DD + d];
    pooled[(size_t)g * DD + d] = s;
}

// ---------------------------------------------------------------------------
// Head: logits[g][c] = sum_i pooled[i][g] . fcW[i][:,c] + sum_i fcb[i][c],
// then log_softmax.  One wave per graph.
// ---------------------------------------------------------------------------
__global__ void head_kernel(const float* __restrict__ pooled,
                            const float* __restrict__ fcW,
                            const float* __restrict__ fcb,
                            float* __restrict__ out) {
    int g    = blockIdx.x;
    int lane = threadIdx.x;                              // 32 threads
    float logits[CC];
    #pragma unroll
    for (int c = 0; c < CC; ++c) logits[c] = 0.0f;

    for (int i = 0; i < LL + 1; ++i) {
        #pragma unroll
        for (int j = 0; j < 4; ++j) {
            int d = lane + 32 * j;
            float p = pooled[((size_t)i * GG + g) * DD + d];
            const float* w = fcW + ((size_t)i * DD + d) * CC;
            #pragma unroll
            for (int c = 0; c < CC; ++c) logits[c] += p * w[c];
        }
    }
    #pragma unroll
    for (int c = 0; c < CC; ++c) {
        float v = logits[c];
        for (int o = 16; o > 0; o >>= 1) v += __shfl_down(v, o, 32);
        logits[c] = v;
    }
    if (lane == 0) {
        #pragma unroll
        for (int c = 0; c < CC; ++c) {
            float b = 0.0f;
            for (int i = 0; i < LL + 1; ++i) b += fcb[i * CC + c];
            logits[c] += b;
        }
        float mx = logits[0];
        #pragma unroll
        for (int c = 1; c < CC; ++c) mx = fmaxf(mx, logits[c]);
        float se = 0.0f;
        #pragma unroll
        for (int c = 0; c < CC; ++c) se += __expf(logits[c] - mx);
        float lse = mx + __logf(se);
        #pragma unroll
        for (int c = 0; c < CC; ++c) out[(size_t)g * CC + c] = logits[c] - lse;
    }
}

// ---------------------------------------------------------------------------
extern "C" void kernel_launch(void* const* d_in, const int* in_sizes, int n_in,
                              void* d_out, int out_size, void* d_ws, size_t ws_size,
                              hipStream_t stream) {
    const float*     x     = (const float*)d_in[0];
    const long long* eidx  = (const long long*)d_in[1];
    const long long* batch = (const long long*)d_in[2];
    const float*     Wc1   = (const float*)d_in[3];
    const float*     bc1   = (const float*)d_in[4];
    const float*     Wc2   = (const float*)d_in[5];
    const float*     bc2   = (const float*)d_in[6];
    const float*     g_mid = (const float*)d_in[7];
    const float*     b_mid = (const float*)d_in[8];
    const float*     g_out = (const float*)d_in[9];
    const float*     b_out = (const float*)d_in[10];
    const float*     fcW   = (const float*)d_in[11];
    const float*     fcb   = (const float*)d_in[12];
    float*           out   = (float*)d_out;

    const long long* src = eidx;            // edge_index[0]
    const long long* dst = eidx + EE;       // edge_index[1]

    // ---- workspace layout (256B aligned) ----
    char* ws = (char*)d_ws;
    size_t off = 0;
    auto carve = [&](size_t bytes) -> char* {
        char* p = ws + off;
        off += (bytes + 255) & ~(size_t)255;
        return p;
    };
    const size_t ndBytes = (size_t)NN * DD * sizeof(float);
    float*    agg    = (float*)carve(ndBytes);          // also reused as u
    float*    t_mid  = (float*)carve(ndBytes);
    float*    h_a    = (float*)carve(ndBytes);
    float*    h_b    = (float*)carve(ndBytes);
    _Float16* W1t    = (_Float16*)carve((size_t)LL * DD * DD * sizeof(_Float16));
    _Float16* W2t    = (_Float16*)carve((size_t)LL * DD * DD * sizeof(_Float16));
    float*    bnsums = (float*)carve(256 * sizeof(float));
    float*    scale1 = (float*)carve(DD * sizeof(float));
    float*    shift1 = (float*)carve(DD * sizeof(float));
    float*    scale2 = (float*)carve(DD * sizeof(float));
    float*    shift2 = (float*)carve(DD * sizeof(float));
    float*    pooled = (float*)carve((size_t)(LL + 1) * GG * DD * sizeof(float));
    int*      gstart = (int*)carve(GG * sizeof(int));
    int*      gend   = (int*)carve(GG * sizeof(int));
    (void)ws_size; (void)in_sizes; (void)n_in; (void)out_size;

    // ---- one-time prep ----
    wconv_kernel<<<(LL * DD * DD + 255) / 256, 256, 0, stream>>>(Wc1, Wc2, W1t, W2t);
    range_init_kernel<<<(GG + 255) / 256, 256, 0, stream>>>(gstart, gend);
    range_kernel<<<(NN + 255) / 256, 256, 0, stream>>>(batch, gstart, gend);
    pool_kernel<<<GG, DD, 0, stream>>>(x, gstart, gend, pooled);   // outs[0]

    const int aggBlocks   = (EE * 32 + 255) / 256;
    const int gemmBlocks  = (NN / 16 + 3) / 4;                     // 782
    const int statsBlocks = (NN + 127) / 128;                      // 391
    const int ewBlocks    = (NN * DD + 255) / 256;

    const float* h_in = x;
    for (int i = 0; i < LL; ++i) {
        float* h_out = (i & 1) ? h_b : h_a;
        const float*    bias1 = bc1 + i * DD;
        const float*    bias2 = bc2 + i * DD;
        const _Float16* B1t   = W1t + (size_t)i * DD * DD;
        const _Float16* B2t   = W2t + (size_t)i * DD * DD;

        // agg = segment_sum(h[src], dst)
        hipMemsetAsync(agg, 0, ndBytes, stream);
        agg_kernel<<<aggBlocks, 256, 0, stream>>>(h_in, src, dst, agg);

        // t = (h + agg) @ Wc1 + bc1
        gemm_add_kernel<<<gemmBlocks, 128, 0, stream>>>(h_in, agg, B1t, bias1, t_mid);

        // BN stats on t -> scale1/shift1
        hipMemsetAsync(bnsums, 0, 256 * sizeof(float), stream);
        stats_kernel<<<statsBlocks, 256, 0, stream>>>(t_mid, bnsums);
        bnfin_kernel<<<1, DD, 0, stream>>>(bnsums, g_mid + i * DD, b_mid + i * DD,
                                           scale1, shift1);

        // u = relu(BN(t)) @ Wc2 + bc2   (u reuses agg buffer)
        gemm_bn_kernel<<<gemmBlocks, 128, 0, stream>>>(t_mid, scale1, shift1,
                                                       B2t, bias2, agg);

        // BN stats on u -> scale2/shift2
        hipMemsetAsync(bnsums, 0, 256 * sizeof(float), stream);
        stats_kernel<<<statsBlocks, 256, 0, stream>>>(agg, bnsums);
        bnfin_kernel<<<1, DD, 0, stream>>>(bnsums, g_out + i * DD, b_out + i * DD,
                                           scale2, shift2);

        // h_out = relu(BN(u)); pooled[i+1] = segment_sum(h_out, batch)
        apply_kernel<<<ewBlocks, 256, 0, stream>>>(agg, scale2, shift2, h_out);
        pool_kernel<<<GG, DD, 0, stream>>>(h_out, gstart, gend,
                                           pooled + (size_t)(i + 1) * GG * DD);
        h_in = h_out;
    }

    head_kernel<<<GG, 32, 0, stream>>>(pooled, fcW, fcb, out);
}
